// GolfStudent_49031346651688
// MI455X (gfx1250) — compile-verified, matrix operand order checked by reading, not run on hardware
//
#include <hip/hip_runtime.h>
#include <hip/hip_bf16.h>
#include <math.h>

typedef __attribute__((ext_vector_type(16))) _Float16 v16h;
typedef __attribute__((ext_vector_type(8)))  float    v8f;
typedef __attribute__((ext_vector_type(4)))  float    v4f;
typedef __attribute__((ext_vector_type(4)))  _Float16 v4h;

#define D_MODEL 288
#define D_FF    864
#define NHEAD   8
#define HD      36
#define SEQ     512
#define BATCH   32
#define MTOK    (BATCH * SEQ)   // 16384 tokens

// ---------------------------------------------------------------------------
// Fragment K index for 16-bit 16x32 A/B WMMA operands (CDNA5 ISA 7.12.2):
// lanes 0-15: rows, VGPR v<4 -> K = 2v + 8*half ; v>=4 -> K = 16 + 2(v-4) + 8*half
// ---------------------------------------------------------------------------
__device__ __forceinline__ int frag_k(int v, int half) {
    return (v < 4) ? (2 * v + 8 * half) : (16 + 2 * (v - 4) + 8 * half);
}

// ---------------------------------------------------------------------------
// Generic GEMM: C[M,N] = A[M,K] * W[N,K]^T  (+ R if ADD_RES)
// block = 256 threads (8 waves), block tile 128(M) x 32(N).
// Waves arranged 4(M) x 2(N); each wave computes 32x16 = two 16x16 WMMA tiles
// sharing one B fragment. Global loads are float4 -> cvt -> 8B LDS stores.
// M % 128 == 0, N % 32 == 0, K % 32 == 0 guaranteed by caller.
// ---------------------------------------------------------------------------
template <int ADD_RES>
__global__ __launch_bounds__(256) void gemm_wmma(const float* __restrict__ A,
                                                 const float* __restrict__ W,
                                                 const float* __restrict__ R,
                                                 float* __restrict__ C,
                                                 int M, int N, int K) {
    __shared__ _Float16 As[128][36];
    __shared__ _Float16 Ws[32][36];
    const int tid  = threadIdx.x;
    const int wave = tid >> 5;
    const int lane = tid & 31;
    const int wr   = wave & 3;   // 0..3  (M direction, 32 rows each)
    const int wc   = wave >> 2;  // 0..1  (N direction, 16 cols each)
    const int m0   = blockIdx.y * 128;
    const int n0   = blockIdx.x * 32;
    const int half = lane >> 4;
    const int mn   = lane & 15;

    v8f acc0 = {}, acc1 = {};
    for (int kk = 0; kk < K; kk += 32) {
        __syncthreads();
        // A tile: 128x32 floats = 1024 float4; 4 per thread
        #pragma unroll
        for (int i = 0; i < 4; ++i) {
            int idx = tid + i * 256;
            int r = idx >> 3, c = (idx & 7) * 4;
            v4f f = *(const v4f*)&A[(size_t)(m0 + r) * K + kk + c];
            *(v4h*)&As[r][c] = __builtin_convertvector(f, v4h);
        }
        // W tile: 32x32 floats = 256 float4; 1 per thread
        {
            int r = tid >> 3, c = (tid & 7) * 4;
            v4f f = *(const v4f*)&W[(size_t)(n0 + r) * K + kk + c];
            *(v4h*)&Ws[r][c] = __builtin_convertvector(f, v4h);
        }
        if (kk + 32 < K) {
            // prefetch next K-slab of A (emits global_prefetch_b8)
            int r = tid >> 1;
            __builtin_prefetch(&A[(size_t)(m0 + r) * K + kk + 32], 0, 1);
        }
        __syncthreads();

        v16h a0, a1, b;
        #pragma unroll
        for (int v = 0; v < 8; ++v) {
            int kb = frag_k(v, half);
            b[2 * v]      = Ws[wc * 16 + mn][kb];
            b[2 * v + 1]  = Ws[wc * 16 + mn][kb + 1];
            a0[2 * v]     = As[wr * 32 + mn][kb];
            a0[2 * v + 1] = As[wr * 32 + mn][kb + 1];
            a1[2 * v]     = As[wr * 32 + 16 + mn][kb];
            a1[2 * v + 1] = As[wr * 32 + 16 + mn][kb + 1];
        }
        acc0 = __builtin_amdgcn_wmma_f32_16x16x32_f16(false, a0, false, b,
                                                      (short)0, acc0, false, false);
        acc1 = __builtin_amdgcn_wmma_f32_16x16x32_f16(false, a1, false, b,
                                                      (short)0, acc1, false, false);
    }
    const int n = n0 + wc * 16 + mn;
    #pragma unroll
    for (int r = 0; r < 8; ++r) {
        int m = m0 + wr * 32 + half * 8 + r;   // C layout: vgpr r -> M = r + 8*half
        size_t off0 = (size_t)m * N + n;
        size_t off1 = (size_t)(m + 16) * N + n;
        float v0 = acc0[r], v1 = acc1[r];
        if (ADD_RES) { v0 += R[off0]; v1 += R[off1]; }
        C[off0] = v0;
        C[off1] = v1;
    }
}

// ---------------------------------------------------------------------------
// Attention: one wave per (batch, head, 16-query tile).
// qkv[M,864] layout: q at col h*36, k at 288+h*36, v at 576+h*36.
// K/V staged through LDS with vectorized branch-free loads.
// HD=36 padded to 64 for QK^T, to 48 for PV.
// ---------------------------------------------------------------------------
__global__ __launch_bounds__(32) void attn_wmma(const float* __restrict__ qkv,
                                                float* __restrict__ o) {
    const int qt = blockIdx.x;   // 0..31 query tile
    const int hh = blockIdx.y;   // 0..7
    const int bb = blockIdx.z;   // 0..31
    __shared__ _Float16 q_s[16][64];    // queries, hd padded to 64
    __shared__ _Float16 kv_s[64][64];   // K group (64 keys x 64) / V tile (32 keys x 48)
    __shared__ float    s_s[16][512];
    __shared__ _Float16 p_s[16][512];

    const int tid  = threadIdx.x;
    const int half = tid >> 4;
    const int mn   = tid & 15;
    const int q0   = qt * 16;
    const size_t base = (size_t)bb * SEQ;
    const v4h zero4 = {};

    // ---- stage Q (scaled by 1/sqrt(36) = 1/6), zero-padded to 64 ----
    if (tid < 16) {
        const float* src = &qkv[(base + q0 + tid) * D_FF + hh * HD];
        #pragma unroll
        for (int j = 0; j < 9; ++j) {
            v4f f = *(const v4f*)&src[j * 4];
            f *= (1.0f / 6.0f);
            *(v4h*)&q_s[tid][j * 4] = __builtin_convertvector(f, v4h);
        }
        #pragma unroll
        for (int j = 9; j < 16; ++j) *(v4h*)&q_s[tid][j * 4] = zero4;
    }
    __syncthreads();

    // ---- QK^T in 64-key groups ----
    const int nkt = qt + 1;                 // causal: key tiles 0..qt
    for (int kg = 0; kg < nkt; kg += 4) {
        // stage 64 K rows (keys kg*16 .. kg*16+63, always < SEQ)
        #pragma unroll
        for (int r2 = 0; r2 < 2; ++r2) {
            int r = tid * 2 + r2;           // 0..63
            const float* src = &qkv[(base + kg * 16 + r) * D_FF + D_MODEL + hh * HD];
            #pragma unroll
            for (int j = 0; j < 9; ++j) {
                v4f f = *(const v4f*)&src[j * 4];
                *(v4h*)&kv_s[r][j * 4] = __builtin_convertvector(f, v4h);
            }
            #pragma unroll
            for (int j = 9; j < 16; ++j) *(v4h*)&kv_s[r][j * 4] = zero4;
        }
        __syncthreads();

        const int ktend = (kg + 4 < nkt) ? (kg + 4) : nkt;
        for (int kt = kg; kt < ktend; ++kt) {
            v8f sc = {};
            #pragma unroll
            for (int kk = 0; kk < 64; kk += 32) {
                v16h a, b;
                #pragma unroll
                for (int v = 0; v < 8; ++v) {
                    int kb = frag_k(v, half);
                    a[2 * v]     = q_s[mn][kk + kb];
                    a[2 * v + 1] = q_s[mn][kk + kb + 1];
                    b[2 * v]     = kv_s[(kt - kg) * 16 + mn][kk + kb];
                    b[2 * v + 1] = kv_s[(kt - kg) * 16 + mn][kk + kb + 1];
                }
                sc = __builtin_amdgcn_wmma_f32_16x16x32_f16(false, a, false, b,
                                                            (short)0, sc, false, false);
            }
            #pragma unroll
            for (int r = 0; r < 8; ++r) {
                int m   = half * 8 + r;
                int key = kt * 16 + mn;
                s_s[m][key] = (key <= q0 + m) ? sc[r] : -3.0e38f;
            }
        }
        __syncthreads();
    }

    // ---- softmax: 2 threads per row, combine with shfl_xor (wave32) ----
    const int row = tid >> 1;
    const int ph  = tid & 1;
    const int Lv  = nkt * 16;
    const int Lp  = (Lv + 31) & ~31;
    float mx = -3.0e38f;
    for (int c = ph; c < Lv; c += 2) mx = fmaxf(mx, s_s[row][c]);
    mx = fmaxf(mx, __shfl_xor(mx, 1, 32));
    float sum = 0.0f;
    for (int c = ph; c < Lv; c += 2) sum += __expf(s_s[row][c] - mx);
    sum += __shfl_xor(sum, 1, 32);
    float inv = 1.0f / sum;
    for (int c = ph; c < Lv; c += 2)
        p_s[row][c] = (_Float16)(__expf(s_s[row][c] - mx) * inv);
    for (int c = Lv + ph; c < Lp; c += 2)
        p_s[row][c] = (_Float16)0.0f;    // zero-pad tail tile
    __syncthreads();

    // ---- PV: M=16 queries, K=keys (Lp), N=HD padded to 48; V staged in LDS ----
    v8f acc[3] = {{}, {}, {}};
    for (int kk = 0; kk < Lp; kk += 32) {
        __syncthreads();
        // stage 32 V rows (keys kk..kk+31), cols 0..47 (36 valid + 12 zero)
        {
            const float* src = &qkv[(base + kk + tid) * D_FF + 2 * D_MODEL + hh * HD];
            #pragma unroll
            for (int j = 0; j < 9; ++j) {
                v4f f = *(const v4f*)&src[j * 4];
                *(v4h*)&kv_s[tid][j * 4] = __builtin_convertvector(f, v4h);
            }
            #pragma unroll
            for (int j = 9; j < 12; ++j) *(v4h*)&kv_s[tid][j * 4] = zero4;
        }
        __syncthreads();

        #pragma unroll
        for (int nt = 0; nt < 3; ++nt) {
            v16h a, b;
            #pragma unroll
            for (int v = 0; v < 8; ++v) {
                int kb = frag_k(v, half);
                a[2 * v]     = p_s[mn][kk + kb];
                a[2 * v + 1] = p_s[mn][kk + kb + 1];
                b[2 * v]     = kv_s[kb][nt * 16 + mn];
                b[2 * v + 1] = kv_s[kb + 1][nt * 16 + mn];
            }
            acc[nt] = __builtin_amdgcn_wmma_f32_16x16x32_f16(false, a, false, b,
                                                             (short)0, acc[nt], false, false);
        }
    }
    #pragma unroll
    for (int nt = 0; nt < 3; ++nt) {
        const int col = nt * 16 + mn;
        #pragma unroll
        for (int r = 0; r < 8; ++r) {
            int m = half * 8 + r;
            if (col < HD)
                o[(base + q0 + m) * D_MODEL + hh * HD + col] = acc[nt][r];
        }
    }
}

// ---------------------------------------------------------------------------
// RMSNorm: one wave per token (288 = 9 * 32)
// ---------------------------------------------------------------------------
__global__ __launch_bounds__(256) void rmsnorm_k(const float* __restrict__ X,
                                                 const float* __restrict__ scale,
                                                 float* __restrict__ Y) {
    const int wave  = threadIdx.x >> 5;
    const int lane  = threadIdx.x & 31;
    const int token = blockIdx.x * 8 + wave;
    const float* x = X + (size_t)token * D_MODEL;
    float vals[9];
    float ss = 0.0f;
    #pragma unroll
    for (int i = 0; i < 9; ++i) {
        vals[i] = x[lane + 32 * i];
        ss += vals[i] * vals[i];
    }
    #pragma unroll
    for (int off = 16; off > 0; off >>= 1) ss += __shfl_xor(ss, off, 32);
    float norm = sqrtf(ss / (float)D_MODEL);
    float inv  = 1.0f / (norm + 1e-6f);
    float* y = Y + (size_t)token * D_MODEL;
    #pragma unroll
    for (int i = 0; i < 9; ++i)
        y[lane + 32 * i] = scale[lane + 32 * i] * vals[i] * inv;
}

// ---------------------------------------------------------------------------
// Embedding + positional
// ---------------------------------------------------------------------------
__global__ void embed_k(const int* __restrict__ x, const float* __restrict__ emb,
                        const float* __restrict__ pos, float* __restrict__ h) {
    size_t i = (size_t)blockIdx.x * blockDim.x + threadIdx.x;
    int d = (int)(i % D_MODEL);
    size_t t = i / D_MODEL;
    int l = (int)(t % SEQ);
    h[i] = emb[(size_t)x[t] * D_MODEL + d] + pos[(size_t)l * D_MODEL + d];
}

// ---------------------------------------------------------------------------
// Recurrence scan: one thread per (batch, channel). Fuses both sigmoids.
// gvf[M,864]: g at 0, v at 288, f at 576.  out gh[M,288] = g * h_seq
// ---------------------------------------------------------------------------
__global__ void scan_k(const float* __restrict__ gvf, float* __restrict__ gh) {
    int idx = blockIdx.x * blockDim.x + threadIdx.x;   // < 32*288
    int d = idx % D_MODEL;
    int b = idx / D_MODEL;
    float hstate = 0.0f;
    for (int l = 0; l < SEQ; ++l) {
        size_t row = ((size_t)b * SEQ + l) * D_FF;
        float graw = gvf[row + d];
        float v    = gvf[row + D_MODEL + d];
        float fraw = gvf[row + 2 * D_MODEL + d];
        float f = 1.0f / (1.0f + __expf(-(fraw - 1.0f)));
        float g = 1.0f / (1.0f + __expf(-graw));
        hstate = f * hstate + (1.0f - f) * v;
        gh[((size_t)b * SEQ + l) * D_MODEL + d] = g * hstate;
    }
}

// ---------------------------------------------------------------------------
// SiLU(gate) * up, in place into gate
// ---------------------------------------------------------------------------
__global__ void silu_mul_k(float* __restrict__ g, const float* __restrict__ u, size_t n) {
    size_t i = (size_t)blockIdx.x * blockDim.x + threadIdx.x;
    if (i < n) {
        float x = g[i];
        g[i] = (x / (1.0f + __expf(-x))) * u[i];
    }
}

// ---------------------------------------------------------------------------
static inline void launch_gemm(const float* A, const float* W, const float* R,
                               float* C, int M, int N, int K, int add_res,
                               hipStream_t s) {
    dim3 grid(N / 32, M / 128);
    if (add_res)
        gemm_wmma<1><<<grid, 256, 0, s>>>(A, W, R, C, M, N, K);
    else
        gemm_wmma<0><<<grid, 256, 0, s>>>(A, W, R, C, M, N, K);
}

extern "C" void kernel_launch(void* const* d_in, const int* in_sizes, int n_in,
                              void* d_out, int out_size, void* d_ws, size_t ws_size,
                              hipStream_t stream) {
    const int*   x          = (const int*)  d_in[0];
    const float* emb        = (const float*)d_in[1];
    const float* pos_emb    = (const float*)d_in[2];
    const float* attn_qkv   = (const float*)d_in[3];
    const float* attn_out   = (const float*)d_in[4];
    const float* attn_norm  = (const float*)d_in[5];
    const float* rec_in     = (const float*)d_in[6];
    const float* rec_out    = (const float*)d_in[7];
    const float* rec_norm   = (const float*)d_in[8];
    const float* ffn_gate   = (const float*)d_in[9];
    const float* ffn_up     = (const float*)d_in[10];
    const float* ffn_down   = (const float*)d_in[11];
    const float* ffn_norm   = (const float*)d_in[12];
    const float* final_norm = (const float*)d_in[13];
    float* out = (float*)d_out;

    // workspace layout (floats)
    float* h  = (float*)d_ws;                   // [MTOK, 288]
    float* xn = h  + (size_t)MTOK * D_MODEL;    // [MTOK, 288]  (also o / gh)
    float* b0 = xn + (size_t)MTOK * D_MODEL;    // [MTOK, 864]  (qkv / gvf / gate)
    float* b1 = b0 + (size_t)MTOK * D_FF;       // [MTOK, 864]  (up)

    // 1. embedding
    embed_k<<<(MTOK * D_MODEL) / 256, 256, 0, stream>>>(x, emb, pos_emb, h);

    int ai = 0, ri = 0;
    for (int i = 0; i < 14; ++i) {
        const bool is_attn = ((i + 1) % 3 == 0);   // layers 2,5,8,11
        if (is_attn) {
            rmsnorm_k<<<MTOK / 8, 256, 0, stream>>>(h, attn_norm + (size_t)ai * D_MODEL, xn);
            launch_gemm(xn, attn_qkv + (size_t)ai * D_FF * D_MODEL, nullptr, b0,
                        MTOK, D_FF, D_MODEL, 0, stream);
            attn_wmma<<<dim3(SEQ / 16, NHEAD, BATCH), 32, 0, stream>>>(b0, xn);
            launch_gemm(xn, attn_out + (size_t)ai * D_MODEL * D_MODEL, h, h,
                        MTOK, D_MODEL, D_MODEL, 1, stream);
            ++ai;
        } else {
            rmsnorm_k<<<MTOK / 8, 256, 0, stream>>>(h, rec_norm + (size_t)ri * D_MODEL, xn);
            launch_gemm(xn, rec_in + (size_t)ri * D_FF * D_MODEL, nullptr, b0,
                        MTOK, D_FF, D_MODEL, 0, stream);
            scan_k<<<(BATCH * D_MODEL) / 256, 256, 0, stream>>>(b0, xn);
            launch_gemm(xn, rec_out + (size_t)ri * D_MODEL * D_MODEL, h, h,
                        MTOK, D_MODEL, D_MODEL, 1, stream);
            ++ri;
        }
        // FFN
        rmsnorm_k<<<MTOK / 8, 256, 0, stream>>>(h, ffn_norm + (size_t)i * D_MODEL, xn);
        launch_gemm(xn, ffn_gate + (size_t)i * D_FF * D_MODEL, nullptr, b0,
                    MTOK, D_FF, D_MODEL, 0, stream);
        launch_gemm(xn, ffn_up + (size_t)i * D_FF * D_MODEL, nullptr, b1,
                    MTOK, D_FF, D_MODEL, 0, stream);
        silu_mul_k<<<((size_t)MTOK * D_FF) / 256, 256, 0, stream>>>(b0, b1, (size_t)MTOK * D_FF);
        launch_gemm(b0, ffn_down + (size_t)i * D_MODEL * D_FF, h, h,
                    MTOK, D_MODEL, D_FF, 1, stream);
    }

    // final norm + tied-embedding logits [MTOK, 4096]
    rmsnorm_k<<<MTOK / 8, 256, 0, stream>>>(h, final_norm, xn);
    launch_gemm(xn, emb, nullptr, out, MTOK, 4096, D_MODEL, 0, stream);
}
